// BiLSTM_CRF_16922171146775
// MI455X (gfx1250) — matrix operand order repaired and checked by reading, not compiled, not run on hardware
//
#include <hip/hip_runtime.h>
#include <hip/hip_bf16.h>
#include <math.h>

// ---------------------------------------------------------------------------
// BiLSTM-CRF forward for MI455X (gfx1250, wave32, WMMA bf16 16x16x32).
//   V=50000 E=300 H=512 Hh=256 B=128 S=128 T=16
//   K1 convert/pad weights to bf16 (E padded 300->320), fuse biases
//   K2 gather embeddings -> x_bf [16384,320] bf16
//   K3 gi GEMM with async-to-LDS A-tile staging; gi stored as raw WMMA
//      C-fragments in bf16 (coalesced 16B/lane)
//   K4 persistent recurrent kernel: 1 WG (512 thr = 16 waves) per direction;
//      h double-buffered in 128KB dynamic LDS; c in global frag scratch;
//      gates fused in-register with native v_tanh_f32
//   K5 projection GEMM (K=512,N=16), 4 m-tiles/wave
//   K6 Viterbi + backtrace (T=16)
// ---------------------------------------------------------------------------

typedef __attribute__((ext_vector_type(16))) __bf16 v16bf;
typedef __attribute__((ext_vector_type(8)))  __bf16 v8bf;
typedef __attribute__((ext_vector_type(8)))  float  v8f;
typedef int async_v4i __attribute__((vector_size(16)));   // builtin's pointee type

#define NEGV      -10000.0f
#define START_TAG 14
#define STOP_TAG  15

// workspace byte offsets (all 32B aligned)
#define OFF_XBF   0UL          // 16384*320 bf16            = 10485760
#define OFF_WIH   10485760UL   // 2*1024*320 bf16           =  1310720
#define OFF_WHH   11796480UL   // 2*1024*256 bf16           =  1048576
#define OFF_WOUT  12845056UL   // 16*512 bf16               =    16384
#define OFF_BIAS  12861440UL   // 2*1024 f32                =     8192
#define OFF_GI    12869632UL   // 2*1024*64 tiles *512B     = 67108864
#define OFF_CFRAG 79978496UL   // 2*8*16*256 f32            =   262144
#define OFF_HALL  80109568UL   // 2*16384*256 bf16          = 16777216
#define OFF_FEATS 96886784UL   // 16384*16 f32              =  1048576
#define OFF_BP    97935360UL   // 128*128*16 u8             =   262144

#if defined(__AMDGCN__) && __has_builtin(__builtin_amdgcn_global_load_async_to_lds_b128) && __has_builtin(__builtin_amdgcn_s_wait_asynccnt)
#define HAS_ASYNC_LDS 1
#endif

__device__ __forceinline__ float fast_tanh(float x) {
#if defined(__gfx1250__) && __has_builtin(__builtin_amdgcn_tanhf)
  return __builtin_amdgcn_tanhf(x);            // v_tanh_f32
#else
  return tanhf(x);
#endif
}
__device__ __forceinline__ float fast_sig(float x) {
  return 0.5f * fast_tanh(0.5f * x) + 0.5f;    // sigmoid via v_tanh_f32
}

// A-fragment (16x32 bf16): lane<16 rows m, K runs {0..15}; lane>=16 runs {16..31}
__device__ __forceinline__ v16bf ldA(const __bf16* p) {
  v8bf lo = *(const v8bf*)p;
  v8bf hi = *(const v8bf*)(p + 8);
  return __builtin_shufflevector(lo, hi, 0,1,2,3,4,5,6,7,8,9,10,11,12,13,14,15);
}

__device__ __forceinline__ v8f wmma_bf16(v16bf a, v16bf b, v8f c) {
  return __builtin_amdgcn_wmma_f32_16x16x32_bf16(false, a, false, b, (short)0, c, false, false);
}

// ---------------------------------------------------------------- K1: weights
__global__ void k_convert_weights(const float* __restrict__ Wih_f,
                                  const float* __restrict__ Wih_b,
                                  const float* __restrict__ Whh_f,
                                  const float* __restrict__ Whh_b,
                                  const float* __restrict__ bih_f,
                                  const float* __restrict__ bhh_f,
                                  const float* __restrict__ bih_b,
                                  const float* __restrict__ bhh_b,
                                  const float* __restrict__ W_out,
                                  char* __restrict__ ws) {
  __bf16* wih  = (__bf16*)(ws + OFF_WIH);
  __bf16* whh  = (__bf16*)(ws + OFF_WHH);
  __bf16* wout = (__bf16*)(ws + OFF_WOUT);
  float*  bias = (float*)(ws + OFF_BIAS);
  int i = blockIdx.x * blockDim.x + threadIdx.x;
  if (i < 2 * 1024 * 320) {   // Wih, pad K 300->320 with zeros
    int dir = i / (1024 * 320), rem = i % (1024 * 320);
    int n = rem / 320, k = rem % 320;
    const float* W = dir ? Wih_b : Wih_f;
    wih[i] = (k < 300) ? (__bf16)W[n * 300 + k] : (__bf16)0.0f;
  }
  if (i < 2 * 1024 * 256) {   // Whh
    int dir = i / (1024 * 256), rem = i % (1024 * 256);
    const float* W = dir ? Whh_b : Whh_f;
    whh[i] = (__bf16)W[rem];
  }
  if (i < 16 * 512) wout[i] = (__bf16)W_out[i];
  if (i < 2 * 1024) {
    int dir = i >> 10, j = i & 1023;
    bias[i] = dir ? (bih_b[j] + bhh_b[j]) : (bih_f[j] + bhh_f[j]);
  }
}

// ---------------------------------------------------------------- K2: gather
__global__ void k_gather(const int* __restrict__ sentence,
                         const float* __restrict__ emb,
                         char* __restrict__ ws) {
  __bf16* xbf = (__bf16*)(ws + OFF_XBF);
  int row = blockIdx.x;          // row = s*128 + b, 16384 rows
  int k   = threadIdx.x;         // 0..319
  int s = row >> 7, b = row & 127;
  int idx = sentence[b * 128 + s];
  float v = (k < 300) ? emb[(size_t)idx * 300 + k] : 0.0f;
  xbf[(size_t)row * 320 + k] = (__bf16)v;
}

// ---------------------------------------------------------------- K3: gi GEMM
// [16384 x 320] x [320 x 1024] per direction. Block: 32 rows x 512 cols.
// A-tile (32x320 bf16 = 20KB) staged into LDS via async load-to-LDS; each
// wave computes 2 m-tiles x 4 n-tiles so every B fragment feeds 2 WMMAs.
__global__ void __launch_bounds__(256) k_input_gemm(char* __restrict__ ws) {
  const __bf16* xbf  = (const __bf16*)(ws + OFF_XBF);
  const __bf16* wih  = (const __bf16*)(ws + OFF_WIH);
  const float*  bias = (const float*)(ws + OFF_BIAS);
  __bf16*       gi   = (__bf16*)(ws + OFF_GI);

  __shared__ __bf16 xtile[32 * 320];    // 20480 B

  int dir  = blockIdx.y;
  int bx   = blockIdx.x;              // 0..1023
  int mp   = bx >> 1;                 // 0..511 (row-pair of 16-tiles)
  int wave = threadIdx.x >> 5;        // 0..7
  int ngrp = (bx & 1) * 8 + wave;     // 0..15
  int lane = threadIdx.x & 31;
  int half = lane >> 4;
  int lcol = lane & 15;
  int n0   = ngrp * 64;

  // ---- stage A tile: rows mp*32 .. mp*32+31, contiguous 20480B region
  {
    const __bf16* src = xbf + (size_t)mp * 32 * 320;
#ifdef HAS_ASYNC_LDS
    for (int c = threadIdx.x; c < 1280; c += 256) {   // 1280 x 16B chunks
      __builtin_amdgcn_global_load_async_to_lds_b128(
          (__attribute__((address_space(1))) async_v4i*)(src + c * 8),
          (__attribute__((address_space(3))) async_v4i*)(xtile + c * 8), 0, 0);
    }
    __builtin_amdgcn_s_wait_asynccnt(0);
#else
    for (int c = threadIdx.x; c < 1280; c += 256)
      *(v8bf*)(xtile + c * 8) = *(const v8bf*)(src + c * 8);
#endif
  }
  __syncthreads();

  const __bf16* wihd = wih + (size_t)dir * 1024 * 320;

  v8f acc[2][4];
  for (int r = 0; r < 2; ++r)
    for (int nt = 0; nt < 4; ++nt) acc[r][nt] = (v8f)0.0f;

  for (int kk = 0; kk < 10; ++kk) {
    int abase = kk * 32 + (half ? 16 : 0);
    v16bf a0 = ldA(xtile + (size_t)(lcol)      * 320 + abase);
    v16bf a1 = ldA(xtile + (size_t)(16 + lcol) * 320 + abase);
    int kb = kk * 32 + (half ? 16 : 0);
    for (int nt = 0; nt < 4; ++nt) {
      int col = n0 + nt * 16 + lcol;
      v16bf b = *(const v16bf*)(wihd + (size_t)col * 320 + kb);
      acc[0][nt] = wmma_bf16(a0, b, acc[0][nt]);
      acc[1][nt] = wmma_bf16(a1, b, acc[1][nt]);
    }
  }
  // epilogue: + bias, store bf16 C-fragments (16B per lane, coalesced)
  for (int r = 0; r < 2; ++r) {
    int tile_m = mp * 2 + r;
    for (int nt = 0; nt < 4; ++nt) {
      int col = n0 + nt * 16 + lcol;
      float bv = bias[dir * 1024 + col];
      v8bf outb;
      for (int v = 0; v < 8; ++v) outb[v] = (__bf16)(acc[r][nt][v] + bv);
      size_t tile = (size_t)tile_m * 64 + (ngrp * 4 + nt);
      *(v8bf*)(gi + (size_t)dir * 16777216 + tile * 256 + lane * 8) = outb;
    }
  }
}

// ---------------------------------------------------------------- K4: LSTM
// One workgroup per direction, 512 threads = 16 waves. Wave w owns hidden
// col-tile jt=w (16 hidden units) and all 4 gates; row-tiles processed in
// pairs so each Whh fragment load feeds 2 WMMAs. h double-buffered in
// dynamic LDS (2*128*256 bf16 = 128KB); c state in global frag scratch.
__global__ void __launch_bounds__(512) k_lstm(const float* __restrict__ h0,
                                              const float* __restrict__ c0,
                                              char* __restrict__ ws) {
  extern __shared__ char smem[];
  __bf16* hb = (__bf16*)smem;                    // [2][128][256]
  const __bf16* whh   = (const __bf16*)(ws + OFF_WHH);
  const __bf16* gi    = (const __bf16*)(ws + OFF_GI);
  float*        cfrag = (float*)(ws + OFF_CFRAG);
  __bf16*       hall  = (__bf16*)(ws + OFF_HALL);

  int dir  = blockIdx.x;
  int tid  = threadIdx.x;
  int jt   = tid >> 5;       // 0..15 hidden col-tile
  int lane = tid & 31;
  int half = lane >> 4;
  int lcol = lane & 15;
  int j    = jt * 16 + lcol; // hidden unit owned by this lane

  const __bf16* whhd = whh + (size_t)dir * 1024 * 256;

  // preload h0 into buffer 0
  for (int i = tid; i < 128 * 256; i += 512)
    hb[i] = (__bf16)h0[dir * 32768 + i];
  // init c fragments from c0 (accumulator layout)
  for (int rt = 0; rt < 8; ++rt) {
    float* cp = cfrag + ((size_t)(dir * 8 + rt) * 16 + jt) * 256 + lane * 8;
    for (int v = 0; v < 8; ++v) {
      int m = rt * 16 + v + 8 * half;
      cp[v] = c0[dir * 32768 + m * 256 + j];
    }
  }
  __syncthreads();

  for (int t = 0; t < 128; ++t) {
    int s_in = dir ? (127 - t) : t;
    int rb = t & 1;
    const __bf16* hrd = hb + (size_t)rb * 128 * 256;
    __bf16*       hwr = hb + (size_t)(rb ^ 1) * 128 * 256;

    for (int rtp = 0; rtp < 4; ++rtp) {
      int rt0 = rtp * 2;
      // init accumulators from gi fragments
      v8f acc[2][4];
      for (int r = 0; r < 2; ++r)
        for (int G = 0; G < 4; ++G) {
          const __bf16* gp = gi + (size_t)dir * 16777216 +
              (((size_t)s_in * 8 + rt0 + r) * 64 + (G * 16 + jt)) * 256 + lane * 8;
          v8bf g8 = *(const v8bf*)gp;
          for (int v = 0; v < 8; ++v) acc[r][G][v] = (float)g8[v];
        }
      // K loop over Hh=256; each B fragment feeds both row-tiles
      for (int kk = 0; kk < 8; ++kk) {
        int abase = kk * 32 + (half ? 16 : 0);
        v16bf a0 = ldA(hrd + (size_t)(rt0 * 16 + lcol)       * 256 + abase);
        v16bf a1 = ldA(hrd + (size_t)((rt0 + 1) * 16 + lcol) * 256 + abase);
        int kb = kk * 32 + (half ? 16 : 0);
        for (int G = 0; G < 4; ++G) {
          const __bf16* bp = whhd + (size_t)(G * 256 + j) * 256 + kb;
          if (kk + 1 < 8) __builtin_prefetch(bp + 32, 0, 1);
          v16bf b = *(const v16bf*)bp;
          acc[0][G] = wmma_bf16(a0, b, acc[0][G]);
          acc[1][G] = wmma_bf16(a1, b, acc[1][G]);
        }
      }
      // fused gate nonlinearity on accumulator layout (native v_tanh_f32)
      for (int r = 0; r < 2; ++r) {
        int rt = rt0 + r;
        float* cp = cfrag + ((size_t)(dir * 8 + rt) * 16 + jt) * 256 + lane * 8;
        v8f cold = *(v8f*)cp;
        for (int v = 0; v < 8; ++v) {
          float ig = fast_sig(acc[r][0][v]);
          float fg = fast_sig(acc[r][1][v]);
          float gg = fast_tanh(acc[r][2][v]);
          float og = fast_sig(acc[r][3][v]);
          float cn = fg * cold[v] + ig * gg;
          float hv = og * fast_tanh(cn);
          cold[v] = cn;
          int m = rt * 16 + v + 8 * half;
          __bf16 hbf = (__bf16)hv;
          hwr[m * 256 + j] = hbf;                                       // next-step A
          hall[((size_t)dir * 16384 + (size_t)s_in * 128 + m) * 256 + j] = hbf;
        }
        *(v8f*)cp = cold;
      }
    }
    __syncthreads();  // all reads of rb done; write buffer fully written
  }
}

// ---------------------------------------------------------------- K5: feats
// [16384 x 512] x [512 x 16]; 8 waves/block, 4 m-tiles per wave so each
// W_out fragment load feeds 4 WMMAs.
__global__ void __launch_bounds__(256) k_proj(const float* __restrict__ b_out,
                                              char* __restrict__ ws) {
  const __bf16* hall = (const __bf16*)(ws + OFF_HALL);
  const __bf16* wout = (const __bf16*)(ws + OFF_WOUT);
  float*        fts  = (float*)(ws + OFF_FEATS);

  int wave = threadIdx.x >> 5;
  int tm0  = (blockIdx.x * 8 + wave) * 4;   // 0..1020
  int lane = threadIdx.x & 31;
  int half = lane >> 4;
  int lcol = lane & 15;

  v8f acc[4];
  for (int mt = 0; mt < 4; ++mt) acc[mt] = (v8f)0.0f;

  for (int kk = 0; kk < 16; ++kk) {
    int dirp = kk >> 3, kk8 = kk & 7;
    int abase = kk8 * 32 + (half ? 16 : 0);
    int kb = kk * 32 + (half ? 16 : 0);
    v16bf b = *(const v16bf*)(wout + (size_t)lcol * 512 + kb);
    for (int mt = 0; mt < 4; ++mt) {
      int arow = (tm0 + mt) * 16 + lcol;
      v16bf a = ldA(hall + ((size_t)dirp * 16384 + arow) * 256 + abase);
      acc[mt] = wmma_bf16(a, b, acc[mt]);
    }
  }
  float bv = b_out[lcol];
  for (int mt = 0; mt < 4; ++mt) {
    for (int v = 0; v < 8; ++v) {
      int r = (tm0 + mt) * 16 + v + 8 * half;   // r = s*128 + b
      int s = r >> 7, b = r & 127;
      fts[((size_t)b * 128 + s) * 16 + lcol] = acc[mt][v] + bv;
    }
  }
}

// ---------------------------------------------------------------- K6: viterbi
// block = 256 threads = 16 batches x 16 tags
__global__ void k_viterbi(const float* __restrict__ trans,
                          char* __restrict__ ws,
                          float* __restrict__ out) {
  const float*   fts = (const float*)(ws + OFF_FEATS);
  unsigned char* bp  = (unsigned char*)(ws + OFF_BP);

  __shared__ float fv[2][16][16];
  __shared__ float tr[256];
  __shared__ float term[16][16];

  int tid = threadIdx.x;
  int bl  = tid >> 4;         // local batch 0..15
  int nxt = tid & 15;         // tag
  int b   = blockIdx.x * 16 + bl;

  tr[tid] = trans[tid];
  fv[0][bl][nxt] = (nxt == START_TAG) ? 0.0f : NEGV;
  __syncthreads();

  for (int s = 0; s < 128; ++s) {
    int rb = s & 1;
    float best = -3.4e38f;
    int arg = 0;
    for (int p = 0; p < 16; ++p) {
      float v = fv[rb][bl][p] + tr[nxt * 16 + p];
      if (v > best) { best = v; arg = p; }
    }
    fv[rb ^ 1][bl][nxt] = best + fts[((size_t)b * 128 + s) * 16 + nxt];
    bp[((size_t)s * 128 + b) * 16 + nxt] = (unsigned char)arg;
    __syncthreads();
  }
  // termination at buffer (128&1)==0
  term[bl][nxt] = fv[0][bl][nxt] + tr[STOP_TAG * 16 + nxt];
  __syncthreads();
  if (nxt == 0) {
    float best = -3.4e38f;
    int arg = 0;
    for (int p = 0; p < 16; ++p)
      if (term[bl][p] > best) { best = term[bl][p]; arg = p; }
    out[b] = best;
    int tag = arg;
    for (int s = 127; s >= 0; --s) {
      out[128 + (size_t)b * 128 + s] = (float)tag;
      tag = bp[((size_t)s * 128 + b) * 16 + tag];
    }
  }
}

// ---------------------------------------------------------------------------
extern "C" void kernel_launch(void* const* d_in, const int* in_sizes, int n_in,
                              void* d_out, int out_size, void* d_ws, size_t ws_size,
                              hipStream_t stream) {
  const int*   sentence = (const int*)d_in[0];
  const float* emb      = (const float*)d_in[1];
  const float* Wih_f    = (const float*)d_in[2];
  const float* Whh_f    = (const float*)d_in[3];
  const float* bih_f    = (const float*)d_in[4];
  const float* bhh_f    = (const float*)d_in[5];
  const float* Wih_b    = (const float*)d_in[6];
  const float* Whh_b    = (const float*)d_in[7];
  const float* bih_b    = (const float*)d_in[8];
  const float* bhh_b    = (const float*)d_in[9];
  const float* W_out    = (const float*)d_in[10];
  const float* b_out    = (const float*)d_in[11];
  const float* trans    = (const float*)d_in[12];
  const float* h0       = (const float*)d_in[13];
  const float* c0       = (const float*)d_in[14];
  char* ws  = (char*)d_ws;
  float* out = (float*)d_out;

  k_convert_weights<<<2560, 256, 0, stream>>>(Wih_f, Wih_b, Whh_f, Whh_b,
                                              bih_f, bhh_f, bih_b, bhh_b,
                                              W_out, ws);
  k_gather<<<16384, 320, 0, stream>>>(sentence, emb, ws);
  k_input_gemm<<<dim3(1024, 2), 256, 0, stream>>>(ws);
  k_lstm<<<2, 512, 2 * 128 * 256 * sizeof(__bf16), stream>>>(h0, c0, ws);
  k_proj<<<32, 256, 0, stream>>>(b_out, ws);
  k_viterbi<<<8, 256, 0, stream>>>(trans, ws, out);
}
